// CausalSelfAttention_8177617731886
// MI455X (gfx1250) — matrix operand order repaired and checked by reading, not compiled, not run on hardware
//
#include <hip/hip_runtime.h>
#include <hip/hip_bf16.h>

// ---------------------------------------------------------------------------
// CDNA5 (gfx1250) causal self-attention: bf16 WMMA pipeline, f32 accumulate,
// async global->LDS tile staging (GLOBAL_LOAD_ASYNC_TO_LDS_B128 + ASYNCcnt).
// B=2, T=2048, C=1024, H=16, D=64.
// ---------------------------------------------------------------------------

typedef __bf16 bf16_t;
typedef __attribute__((ext_vector_type(16))) __bf16 v16bf;
typedef __attribute__((ext_vector_type(8)))  __bf16 v8bf;
typedef __attribute__((ext_vector_type(8)))  float  v8f;

#define T_SEQ 2048
#define C_DIM 1024
#define H_N   16
#define D_H   64
#define SCALE_F 0.125f   // 1/sqrt(64)

union FragBF {
  v16bf v;
  v8bf  h[2];
};

__device__ __forceinline__ bf16_t f2bf(float f) {
  unsigned u = __builtin_bit_cast(unsigned, f);
  unsigned r = u + 0x7FFFu + ((u >> 16) & 1u);   // round-to-nearest-even
  return __builtin_bit_cast(bf16_t, (unsigned short)(r >> 16));
}

// Async 16B copy global -> LDS. LDS byte offset is the low 32 bits of the
// generic pointer (ISA: flat aperture check).
__device__ __forceinline__ void async_copy_b128(void* ldsDst, const void* gSrc) {
  unsigned lds = (unsigned)(uintptr_t)ldsDst;
  asm volatile("global_load_async_to_lds_b128 %0, %1, off"
               :: "v"(lds), "v"(gSrc) : "memory");
}
__device__ __forceinline__ void wait_async0() {
  asm volatile("s_wait_asynccnt 0" ::: "memory");
}

// ---------------------------------------------------------------------------
// fp32 -> bf16 elementwise conversion
// ---------------------------------------------------------------------------
__global__ __launch_bounds__(256)
void convert_f32_bf16(const float* __restrict__ in, bf16_t* __restrict__ out, int n) {
  int i = blockIdx.x * 256 + threadIdx.x;
  int stride = gridDim.x * 256;
  for (; i < n; i += stride) out[i] = f2bf(in[i]);
}

// ---------------------------------------------------------------------------
// GEMM: out = A[N,K] @ W[Cout,K]^T  via v_wmma_f32_16x16x32_bf16
// 8 waves/block, each wave: 32 rows x 64 cols (8 WMMA / k-step).
// 64x32 W tile staged in LDS by async copies, shared by all waves.
// MODE 0: bf16 out, [B,H,T,D]   (Q, K projections)
// MODE 1: bf16 out, [B,H,D,T]   (V projection, pre-transposed for PV WMMA)
// MODE 2: f32  out, [N,Cout]    (final output projection)
// ---------------------------------------------------------------------------
template <int MODE>
__global__ __launch_bounds__(256)
void gemm_wmma(const bf16_t* __restrict__ A, const bf16_t* __restrict__ W,
               float* __restrict__ outF, bf16_t* __restrict__ outB) {
  __shared__ bf16_t wTile[64][32];           // [col][k] 4 KB

  const int tid  = threadIdx.x;
  const int lane = tid & 31;
  const int wave = tid >> 5;
  const int lo   = lane & 15;
  const int hi   = lane >> 4;

  const int rowBase = (blockIdx.x * 8 + wave) * 32;   // 32 rows per wave
  const int colBase = blockIdx.y * 64;

  v8f acc[2][4] = {};

  for (int k0 = 0; k0 < C_DIM; k0 += 32) {
    __syncthreads();                      // previous-iteration consumers done
    {
      // async stage 64x32 W tile: 256 threads x 16B
      int r = tid >> 2, ch = tid & 3;
      async_copy_b128(&wTile[r][ch * 8],
                      W + (size_t)(colBase + r) * C_DIM + k0 + ch * 8);
    }
    wait_async0();
    __syncthreads();

    FragBF a0, a1;
    const bf16_t* arow0 = A + (size_t)(rowBase + lo) * C_DIM + k0;
    const bf16_t* arow1 = arow0 + (size_t)16 * C_DIM;
    a0.h[0] = *(const v8bf*)(arow0 + hi * 8);
    a0.h[1] = *(const v8bf*)(arow0 + 16 + hi * 8);
    a1.h[0] = *(const v8bf*)(arow1 + hi * 8);
    a1.h[1] = *(const v8bf*)(arow1 + 16 + hi * 8);
    __builtin_prefetch(arow0 + 64, 0, 0);   // global_prefetch_b8 k-tile two steps ahead
    __builtin_prefetch(arow1 + 64, 0, 0);

#pragma unroll
    for (int c = 0; c < 4; ++c) {
      FragBF b;
      b.h[0] = *(const v8bf*)&wTile[c * 16 + lo][hi * 8];
      b.h[1] = *(const v8bf*)&wTile[c * 16 + lo][16 + hi * 8];
      acc[0][c] = __builtin_amdgcn_wmma_f32_16x16x32_bf16(
          false, a0.v, false, b.v, (short)0, acc[0][c], false, false);
      acc[1][c] = __builtin_amdgcn_wmma_f32_16x16x32_bf16(
          false, a1.v, false, b.v, (short)0, acc[1][c], false, false);
    }
  }

#pragma unroll
  for (int mt = 0; mt < 2; ++mt) {
#pragma unroll
    for (int c = 0; c < 4; ++c) {
#pragma unroll
      for (int r = 0; r < 8; ++r) {
        float v = acc[mt][c][r];
        int m = rowBase + mt * 16 + r + 8 * hi;   // global row (b*T + t)
        int o = colBase + c * 16 + lo;            // output channel
        if (MODE == 2) {
          outF[(size_t)m * C_DIM + o] = v;
        } else {
          int b = m >> 11, t = m & (T_SEQ - 1);
          int h = o >> 6,  d = o & (D_H - 1);
          size_t idx;
          if (MODE == 0) idx = (((size_t)(b * H_N + h)) * T_SEQ + t) * D_H + d;
          else           idx = (((size_t)(b * H_N + h)) * D_H  + d) * T_SEQ + t;
          outB[idx] = f2bf(v);
        }
      }
    }
  }
}

// ---------------------------------------------------------------------------
// Flash attention (causal, online softmax). One block = (b,h) x 128 q rows,
// 4 waves, each wave: 32 query rows (two 16-row subtiles sharing K/V tiles).
// Key tiles of 32 async-staged in LDS.
// Q,K: [B,H,T,D] bf16.  Vt: [B,H,D,T] bf16.  O: [B*T, C] bf16.
// ---------------------------------------------------------------------------
__global__ __launch_bounds__(128)
void flash_attn(const bf16_t* __restrict__ Q, const bf16_t* __restrict__ K,
                const bf16_t* __restrict__ Vt, bf16_t* __restrict__ O) {
  __shared__ bf16_t kTile[32][64];       // [key][d]   4 KB
  __shared__ bf16_t vTile[64][32];       // [d][key]   4 KB
  __shared__ bf16_t pTile[4][16][32];    // per-wave P 4 KB (reused per q-subtile)

  const int tid  = threadIdx.x;
  const int lane = tid & 31;
  const int wave = tid >> 5;
  const int lo   = lane & 15;
  const int hi   = lane >> 4;

  const int bh    = blockIdx.y;                 // b*H + h
  const int qBase = blockIdx.x * 128 + wave * 32;

  const bf16_t* Qp = Q  + (size_t)bh * T_SEQ * D_H;
  const bf16_t* Kp = K  + (size_t)bh * T_SEQ * D_H;
  const bf16_t* Vp = Vt + (size_t)bh * D_H * T_SEQ;

  // Q A-fragments: aq[qt][dc], qt = 16-row subtile, dc = d 0..31 / 32..63
  FragBF aq[2][2];
#pragma unroll
  for (int qt = 0; qt < 2; ++qt) {
    const bf16_t* qrow = Qp + (size_t)(qBase + qt * 16 + lo) * D_H;
    aq[qt][0].h[0] = *(const v8bf*)(qrow + hi * 8);
    aq[qt][0].h[1] = *(const v8bf*)(qrow + 16 + hi * 8);
    aq[qt][1].h[0] = *(const v8bf*)(qrow + 32 + hi * 8);
    aq[qt][1].h[1] = *(const v8bf*)(qrow + 48 + hi * 8);
  }

  v8f oacc[2][4] = {};
  float mrow[2][8], lrow[2][8];
#pragma unroll
  for (int qt = 0; qt < 2; ++qt)
#pragma unroll
    for (int r = 0; r < 8; ++r) { mrow[qt][r] = -1e30f; lrow[qt][r] = 0.0f; }

  const int kEnd = blockIdx.x * 128 + 128;   // keys needed by whole block

  for (int kBase = 0; kBase < kEnd; kBase += 32) {
    __syncthreads();
    {   // async stage K tile: 32 keys x 64 d  (128 thr x 32B)
      int r = tid >> 2, ch = tid & 3;
      const bf16_t* src = Kp + (size_t)(kBase + r) * D_H + ch * 16;
      async_copy_b128(&kTile[r][ch * 16],     src);
      async_copy_b128(&kTile[r][ch * 16 + 8], src + 8);
    }
    {   // async stage V tile: 64 d x 32 keys (already transposed in memory)
      int d = tid >> 1, ch = tid & 1;
      const bf16_t* src = Vp + (size_t)d * T_SEQ + kBase + ch * 16;
      async_copy_b128(&vTile[d][ch * 16],     src);
      async_copy_b128(&vTile[d][ch * 16 + 8], src + 8);
    }
    wait_async0();
    __syncthreads();

    if (kBase <= qBase + 31) {     // wave has at least one unmasked q-subtile
#pragma unroll
      for (int qt = 0; qt < 2; ++qt) {
        const int qtBase = qBase + qt * 16;
        if (kBase > qtBase + 15) continue;   // this q-subtile fully masked

        // ---- scores: S = Q (16x64) x K^T (64x32), two 16-key subtiles ----
        v8f s[2];
#pragma unroll
        for (int st = 0; st < 2; ++st) {
          int key = st * 16 + lo;
          FragBF b0, b1;   // B frag: N=key, K-dim = d
          b0.h[0] = *(const v8bf*)&kTile[key][hi * 8];
          b0.h[1] = *(const v8bf*)&kTile[key][16 + hi * 8];
          b1.h[0] = *(const v8bf*)&kTile[key][32 + hi * 8];
          b1.h[1] = *(const v8bf*)&kTile[key][48 + hi * 8];
          v8f t = {};
          t = __builtin_amdgcn_wmma_f32_16x16x32_bf16(
              false, aq[qt][0].v, false, b0.v, (short)0, t, false, false);
          t = __builtin_amdgcn_wmma_f32_16x16x32_bf16(
              false, aq[qt][1].v, false, b1.v, (short)0, t, false, false);
          s[st] = t;
        }
        // ---- scale + causal mask ----
#pragma unroll
        for (int st = 0; st < 2; ++st)
#pragma unroll
          for (int r = 0; r < 8; ++r) {
            int qg = qtBase + r + 8 * hi;
            int kg = kBase + st * 16 + lo;
            float v = s[st][r] * SCALE_F;
            s[st][r] = (kg <= qg) ? v : -1e30f;
          }
        // ---- online softmax (per-row, reductions in 16-lane halves) ----
#pragma unroll
        for (int r = 0; r < 8; ++r) {
          float mx = fmaxf(s[0][r], s[1][r]);
#pragma unroll
          for (int msk = 8; msk >= 1; msk >>= 1)
            mx = fmaxf(mx, __shfl_xor(mx, msk, 32));
          float mnew  = fmaxf(mrow[qt][r], mx);
          float alpha = __expf(mrow[qt][r] - mnew);
          float p0 = __expf(s[0][r] - mnew);
          float p1 = __expf(s[1][r] - mnew);
          float ps = p0 + p1;
#pragma unroll
          for (int msk = 8; msk >= 1; msk >>= 1)
            ps += __shfl_xor(ps, msk, 32);
          lrow[qt][r] = lrow[qt][r] * alpha + ps;
          mrow[qt][r] = mnew;
          oacc[qt][0][r] *= alpha; oacc[qt][1][r] *= alpha;
          oacc[qt][2][r] *= alpha; oacc[qt][3][r] *= alpha;
          int mm = r + 8 * hi;
          pTile[wave][mm][lo]      = f2bf(p0);
          pTile[wave][mm][16 + lo] = f2bf(p1);
        }
        // ---- O += P (16x32) x V (32x64) ----
        FragBF pa;   // A frag of P: row = lo, K-dim = key'
        pa.h[0] = *(const v8bf*)&pTile[wave][lo][hi * 8];
        pa.h[1] = *(const v8bf*)&pTile[wave][lo][16 + hi * 8];
#pragma unroll
        for (int c = 0; c < 4; ++c) {
          FragBF vb;   // B frag of V: N = d, K-dim = key'
          vb.h[0] = *(const v8bf*)&vTile[c * 16 + lo][hi * 8];
          vb.h[1] = *(const v8bf*)&vTile[c * 16 + lo][16 + hi * 8];
          oacc[qt][c] = __builtin_amdgcn_wmma_f32_16x16x32_bf16(
              false, pa.v, false, vb.v, (short)0, oacc[qt][c], false, false);
        }
      }
    }
  }

  // ---- normalize + store to [B*T, C] bf16 for the output projection ----
  const int b = bh >> 4, h = bh & 15;
#pragma unroll
  for (int qt = 0; qt < 2; ++qt) {
#pragma unroll
    for (int r = 0; r < 8; ++r) {
      float inv = 1.0f / lrow[qt][r];
      size_t n = (size_t)(b * T_SEQ + qBase + qt * 16 + r + 8 * hi);
      size_t base = n * C_DIM + h * D_H;
      O[base + lo]      = f2bf(oacc[qt][0][r] * inv);
      O[base + 16 + lo] = f2bf(oacc[qt][1][r] * inv);
      O[base + 32 + lo] = f2bf(oacc[qt][2][r] * inv);
      O[base + 48 + lo] = f2bf(oacc[qt][3][r] * inv);
    }
  }
}

// ---------------------------------------------------------------------------
// Host launch
// ---------------------------------------------------------------------------
extern "C" void kernel_launch(void* const* d_in, const int* in_sizes, int n_in,
                              void* d_out, int out_size, void* d_ws, size_t ws_size,
                              hipStream_t stream) {
  const float* x  = (const float*)d_in[0];
  const float* wq = (const float*)d_in[1];
  const float* wk = (const float*)d_in[2];
  const float* wv = (const float*)d_in[3];
  const float* wo = (const float*)d_in[4];
  float* out = (float*)d_out;

  const size_t N  = 2ull * T_SEQ;          // 4096 rows
  const size_t NC = N * C_DIM;             // 4,194,304
  const size_t WW = (size_t)C_DIM * C_DIM; // 1,048,576

  char* ws = (char*)d_ws;
  bf16_t* xb  = (bf16_t*)ws;                 ws += NC * 2;
  bf16_t* wqb = (bf16_t*)ws;                 ws += WW * 2;
  bf16_t* wkb = (bf16_t*)ws;                 ws += WW * 2;
  bf16_t* wvb = (bf16_t*)ws;                 ws += WW * 2;
  bf16_t* wob = (bf16_t*)ws;                 ws += WW * 2;
  bf16_t* Qb  = (bf16_t*)ws;                 ws += NC * 2;   // [B,H,T,D]
  bf16_t* Kb  = (bf16_t*)ws;                 ws += NC * 2;   // [B,H,T,D]
  bf16_t* Vb  = (bf16_t*)ws;                 ws += NC * 2;   // [B,H,D,T]
  bf16_t* AOb = (bf16_t*)ws;                 ws += NC * 2;   // [B*T, C]

  // 1) fp32 -> bf16 conversions
  convert_f32_bf16<<<dim3((unsigned)((NC + 255) / 256)), 256, 0, stream>>>(x,  xb,  (int)NC);
  convert_f32_bf16<<<dim3((unsigned)((WW + 255) / 256)), 256, 0, stream>>>(wq, wqb, (int)WW);
  convert_f32_bf16<<<dim3((unsigned)((WW + 255) / 256)), 256, 0, stream>>>(wk, wkb, (int)WW);
  convert_f32_bf16<<<dim3((unsigned)((WW + 255) / 256)), 256, 0, stream>>>(wv, wvb, (int)WW);
  convert_f32_bf16<<<dim3((unsigned)((WW + 255) / 256)), 256, 0, stream>>>(wo, wob, (int)WW);

  // 2) Q/K/V projections (WMMA GEMMs): block tile 256 rows x 64 cols
  dim3 gg((unsigned)(N / 256), (unsigned)(C_DIM / 64));
  gemm_wmma<0><<<gg, 256, 0, stream>>>(xb, wqb, nullptr, Qb);
  gemm_wmma<0><<<gg, 256, 0, stream>>>(xb, wkb, nullptr, Kb);
  gemm_wmma<1><<<gg, 256, 0, stream>>>(xb, wvb, nullptr, Vb);

  // 3) causal flash attention: block = 128 q rows per (b,h)
  flash_attn<<<dim3((unsigned)(T_SEQ / 128), (unsigned)(2 * H_N)), 128, 0, stream>>>(
      Qb, Kb, Vb, AOb);

  // 4) output projection -> fp32 d_out
  gemm_wmma<2><<<gg, 256, 0, stream>>>(AOb, wob, out, nullptr);
}
